// HyenaFilter_29609504539254
// MI455X (gfx1250) — matrix operand order repaired and checked by reading, not compiled, not run on hardware
//
#include <hip/hip_runtime.h>
#include <hip/hip_bf16.h>

// ---------------- problem constants ----------------
#define L_SEQ   8192
#define D_DIM   1024
#define BATCH   4
#define W_FILT  64

// padded layouts
#define XPAD    64                  // zero pad on both ends of x (bf16)
#define LXP     (L_SEQ + 2*XPAD)    // 8320
#define KRP     (L_SEQ + 64)        // reversed k, 64-element zero tail

// conv tiling
#define KBLK    256                 // outer K block staged in LDS
#define XS_W    344                 // x window row: 304 used, padded to 43 x b128
#define KS_W    736                 // kr window: <=719 used, padded to 92 x b128

typedef __attribute__((ext_vector_type(16))) __bf16 v16bf;
typedef __attribute__((ext_vector_type(8)))  __bf16 v8bf;
typedef __attribute__((ext_vector_type(8)))  float  v8f;
typedef int v4i __attribute__((vector_size(4 * sizeof(int))));   // matches builtin param

// ---------------- gfx1250 async-to-LDS plumbing ----------------
#if __has_builtin(__builtin_amdgcn_global_load_async_to_lds_b128)
#define HAVE_ASYNC 1
#else
#define HAVE_ASYNC 0
#endif

#if __has_builtin(__builtin_amdgcn_s_wait_asynccnt)
#define WAIT_ASYNC(n) __builtin_amdgcn_s_wait_asynccnt(n)
#elif HAVE_ASYNC
#define WAIT_ASYNC(n) asm volatile("s_wait_asynccnt %0" :: "i"(n) : "memory")
#else
#define WAIT_ASYNC(n) do {} while (0)
#endif

// ---------------- kernel 1: x -> padded bf16 ----------------
__global__ __launch_bounds__(256) void prep_x_bf16(const float* __restrict__ x,
                                                   __bf16* __restrict__ xbp) {
    size_t i  = (size_t)blockIdx.x * 256 + threadIdx.x;   // over BATCH*D_DIM*LXP
    int    pos = (int)(i % LXP);
    size_t bd  = i / LXP;
    int    m   = pos - XPAD;
    float  v   = (m >= 0 && m < L_SEQ) ? x[bd * L_SEQ + m] : 0.0f;
    xbp[i] = (__bf16)v;
}

// ---------------- kernel 2: implicit filter MLP -> reversed bf16 k ----------------
__global__ __launch_bounds__(256) void hyena_filter(const float* __restrict__ w1,
                                                    const float* __restrict__ b1,
                                                    const float* __restrict__ w2,
                                                    const float* __restrict__ b2,
                                                    const float* __restrict__ w3,
                                                    const float* __restrict__ bias,
                                                    __bf16* __restrict__ kr) {
    __shared__ float h1[W_FILT];
    __shared__ float h2[W_FILT];
    const int l   = blockIdx.x;
    const int tid = threadIdx.x;
    const float t = (float)l / (float)(L_SEQ - 1);

    if (tid < W_FILT) {
        float s1 = w1[tid] + w1[W_FILT + tid] + w1[2 * W_FILT + tid];
        float z  = t * s1 + b1[tid];
        h1[tid]  = z / (1.0f + __expf(-z));              // silu
    }
    __syncthreads();
    if (tid < W_FILT) {
        float acc = b2[tid];
        #pragma unroll 8
        for (int i = 0; i < W_FILT; ++i) acc += h1[i] * w2[i * W_FILT + tid];
        h2[tid] = acc / (1.0f + __expf(-acc));           // silu
    }
    __syncthreads();
    #pragma unroll
    for (int q = 0; q < 4; ++q) {
        int d = tid + 256 * q;                           // 0..1023
        float acc = 0.0f;
        #pragma unroll 8
        for (int i = 0; i < W_FILT; ++i) acc += h2[i] * w3[i * D_DIM + d];
        acc *= __expf(-t * fabsf(bias[d]));              // decay modulation
        kr[(size_t)d * KRP + (L_SEQ - 1 - l)] = (__bf16)acc;   // store reversed
    }
}

__global__ __launch_bounds__(256) void zero_kr_tail(__bf16* __restrict__ kr) {
    int i = blockIdx.x * 256 + threadIdx.x;              // over D_DIM*64
    int d = i >> 6, j = i & 63;
    kr[(size_t)d * KRP + L_SEQ + j] = (__bf16)0.0f;
}

// ---------------- kernel 3: causal Toeplitz conv via WMMA ----------------
// grid: (D_DIM, L_SEQ/512). 256 threads = 8 waves. Wave w owns outputs
// [lbase+64w, lbase+64w+63] as 4 column groups x 4 batches = 16 WMMA columns.
// Double-buffered LDS windows filled by GLOBAL_LOAD_ASYNC_TO_LDS_B128.
__global__ __launch_bounds__(256) void hyena_conv_wmma(const __bf16* __restrict__ xbp,
                                                       const __bf16* __restrict__ kr,
                                                       const float*  __restrict__ x,
                                                       const float*  __restrict__ bias,
                                                       float*        __restrict__ out) {
    const int d     = blockIdx.x;
    const int lbase = blockIdx.y * 512;
    const int tid   = threadIdx.x;
    const int wave  = tid >> 5;
    const int lane  = tid & 31;

    __shared__ alignas(32) __bf16 xs2[2][BATCH][XS_W];
    __shared__ alignas(32) __bf16 ks2[2][KS_W];

    const int l0 = lbase + 64 * wave + 48;               // top output tile of this wave
    const __bf16* krd = kr + (size_t)d * KRP;

    const int hi = (lane >= 16);                         // which K-half this lane holds
    const int Mr = lane & 15;                            // A row / output column index
    const int g  = Mr >> 2;                              // column group -> l-shift 16*g
    const int bb = Mr & 3;                               // batch

    v8f acc;
    #pragma unroll
    for (int j = 0; j < 8; ++j) acc[j] = 0.0f;

    // ---- per-block staging: each thread moves exactly one x-b128 and one k-b128
    const int ex = tid % 172;                            // 172 = 4 rows * 43 xfers
    const int xb = ex / 43, xo = (ex % 43) * 8;          // element offset in row
    const int ek = (tid % 92) * 8;                       // 92 k-window xfers
    const size_t xrowbase = ((size_t)(xb * D_DIM + d)) * LXP + (XPAD - 48);

    auto stage = [&](int m0base, int buf) {
        const int krwin0 = (L_SEQ - 512 - lbase) + m0base;
#if HAVE_ASYNC
        __builtin_amdgcn_global_load_async_to_lds_b128(
            (v4i*)(xbp + xrowbase + m0base + xo), (v4i*)(&xs2[buf][xb][xo]), 0, 0);
        int ki = krwin0 + ek;
        if (ki > KRP - 8) ki = KRP - 8;                  // clamped slots never read
        __builtin_amdgcn_global_load_async_to_lds_b128(
            (v4i*)(krd + ki), (v4i*)(&ks2[buf][ek]), 0, 0);
#else
        for (int e = tid; e < BATCH * XS_W; e += 256) {
            int b = e / XS_W, o = e % XS_W;
            xs2[buf][b][o] = xbp[((size_t)(b * D_DIM + d)) * LXP + (XPAD - 48) + m0base + o];
        }
        for (int e = tid; e < KS_W; e += 256) {
            int idx = krwin0 + e;
            ks2[buf][e] = (idx < KRP) ? krd[idx] : (__bf16)0.0f;
        }
#endif
    };

    const int nblk = (lbase + 512) / KBLK;               // >= 2
    stage(0, 0);                                         // prologue

    for (int blk = 0; blk < nblk; ++blk) {
        const int m0base = blk * KBLK;
        const int cur    = blk & 1;
        const bool more  = (blk + 1 < nblk);
        if (more) stage(m0base + KBLK, cur ^ 1);         // next block in flight
        if (more) { WAIT_ASYNC(2); } else { WAIT_ASYNC(0); }
        __syncthreads();                                 // all waves' window ready

        #pragma unroll
        for (int s = 0; s < KBLK / 32; ++s) {
            const int m0 = m0base + 32 * s;
            if (m0 <= l0 + 15) {                         // wave-uniform: EXEC stays full
                // ---- A: 16x32 bf16 Toeplitz tile of reversed k (per-lane u16 gather;
                //      offsets have lane-dependent parity, inherent to Toeplitz shift)
                v16bf a;
                const int abase = 463 - 64 * wave - Mr + 32 * s;
                const int kkA0  = hi ? 16 : 0;
                const int kkA1  = hi ? 24 : 8;
                #pragma unroll
                for (int j = 0; j < 8; ++j) {
                    a[j]     = ks2[cur][abase + kkA0 + j];
                    a[j + 8] = ks2[cur][abase + kkA1 + j];
                }
                // ---- B: 32x16 bf16, column (g,b) holds x[b,d, m0+kk-16g].
                //      bbase is a multiple of 16 elements -> two aligned ds_load_b128.
                const int bbase = 32 * s + 48 - 16 * g + (hi ? 16 : 0);
                v8bf blo = *(const v8bf*)&xs2[cur][bb][bbase];
                v8bf bhi = *(const v8bf*)&xs2[cur][bb][bbase + 8];
                v16bf bm = __builtin_shufflevector(blo, bhi,
                              0, 1, 2, 3, 4, 5, 6, 7, 8, 9, 10, 11, 12, 13, 14, 15);

                acc = __builtin_amdgcn_wmma_f32_16x16x32_bf16(
                        false, a, false, bm, (short)0, acc, false, false);
            }
        }
        __syncthreads();                                 // done reading buf `cur`
    }

    // ---- store: column (g,bb) is output tile l0-16g; VGPR r -> rows r / r+8
    const int   lout0 = l0 - 16 * g + (hi ? 8 : 0);
    const float bd    = bias[d];
    const size_t obase = ((size_t)bb * D_DIM + d) * L_SEQ;
    #pragma unroll
    for (int r = 0; r < 8; ++r) {
        int l = lout0 + r;
        out[obase + l] = acc[r] + x[obase + l] * bd;     // fused skip: y + x*bias
    }
}

// ---------------- launcher ----------------
extern "C" void kernel_launch(void* const* d_in, const int* in_sizes, int n_in,
                              void* d_out, int out_size, void* d_ws, size_t ws_size,
                              hipStream_t stream) {
    const float* x    = (const float*)d_in[0];
    const float* w1   = (const float*)d_in[1];
    const float* b1   = (const float*)d_in[2];
    const float* w2   = (const float*)d_in[3];
    const float* b2   = (const float*)d_in[4];
    const float* w3   = (const float*)d_in[5];
    const float* bias = (const float*)d_in[6];
    float* out = (float*)d_out;

    // workspace layout: [xbp bf16: B*D*LXP][kr bf16: D*KRP]
    const size_t xbp_elems = (size_t)BATCH * D_DIM * LXP;
    __bf16* xbp = (__bf16*)d_ws;
    __bf16* kr  = (__bf16*)((char*)d_ws + xbp_elems * sizeof(__bf16));

    prep_x_bf16<<<dim3((unsigned)(xbp_elems / 256)), dim3(256), 0, stream>>>(x, xbp);
    hyena_filter<<<dim3(L_SEQ), dim3(256), 0, stream>>>(w1, b1, w2, b2, w3, bias, kr);
    zero_kr_tail<<<dim3((D_DIM * 64) / 256), dim3(256), 0, stream>>>(kr);
    hyena_conv_wmma<<<dim3(D_DIM, L_SEQ / 512), dim3(256), 0, stream>>>(xbp, kr, x, bias, out);
}
